// Vit_7318624272439
// MI455X (gfx1250) — matrix operand order repaired and checked by reading, not compiled, not run on hardware
//
#include <hip/hip_runtime.h>
#include <math.h>

// ---------------------------------------------------------------------------
// ViT-Base forward for MI455X (gfx1250, wave32, WMMA).
//
// All operands pre-converted to f16 (weights once per launch, activations at
// their producers).  GEMM inner loop: per wave a 32x64 output tile
// (8 accumulators, 8 v_wmma_f32_16x16x32_f16 per K-step, each B fragment
// reused by two A fragments).  The 64x32 B tile is double-buffered in LDS and
// filled with global_load_async_to_lds_b128 (ASYNCcnt, via inline asm),
// overlapping the copy of step k+1 with the WMMAs of step k behind a single
// barrier per iteration.  Tokens padded 197 -> 224; pad keys masked in softmax.
// ---------------------------------------------------------------------------

typedef __attribute__((ext_vector_type(16))) _Float16 v16h;
typedef __attribute__((ext_vector_type(8)))  float    v8f;
typedef _Float16 h16;
typedef long long ll;

union Frag {
    v16h v;
    uint4 q[2];
};

#if defined(__has_builtin)
#  if __has_builtin(__builtin_amdgcn_global_load_async_to_lds_b128)
#    define VIT_ASYNC_LDS 1
#  endif
#endif

// Copy 16 bytes global -> LDS (async via ASYNCcnt when the toolchain knows
// gfx1250 async ops).  The LDS operand is the per-lane LDS byte offset: for
// flat addresses the ISA maps addr[31:0] directly to the LDS offset, so the
// truncated generic pointer to a __shared__ object is exactly what VDST wants.
__device__ __forceinline__ void copy16_g2l(h16* dst, const h16* src) {
#ifdef VIT_ASYNC_LDS
    const unsigned ldsoff = (unsigned)(size_t)dst;
    asm volatile("global_load_async_to_lds_b128 %0, %1, off"
                 :
                 : "v"(ldsoff), "v"(src)
                 : "memory");
#else
    *(uint4*)dst = *(const uint4*)src;
#endif
}

__device__ __forceinline__ void wait_g2l() {
#ifdef VIT_ASYNC_LDS
#  if __has_builtin(__builtin_amdgcn_s_wait_asynccnt)
    __builtin_amdgcn_s_wait_asynccnt(0);
#  else
    asm volatile("s_wait_asynccnt 0x0" ::: "memory");
#  endif
#endif
}

// ---------------------------------------------------------------------------
// Batched f16 WMMA GEMM:  C[z] = act(alpha * A[z] * B[z]^T + bias) (+ R[z])
//   A: (M,K) f16 row-major.  Bw: (N,K) f16 row-major.  R: fp32, C's offsets.
//   C: fp32 (outHalf=0) or f16 (outHalf=1).
// Batch z -> (outer, inner) = (z/innerCount, z%innerCount); operand offset =
// outer*s?o + inner*s?i (element units of that operand's type).
// Requires K%32==0, M%32==0; N guarded per 16-col sub-tile; M guarded per wave.
// Block = 128 threads = 4 waves; block tile = 128(M) x 64(N); wave w owns rows
// [blockIdx.x*128 + w*32, +32).  B tile (64x32 halfs) double-buffered in LDS.
// ---------------------------------------------------------------------------
__launch_bounds__(128)
__global__ void wmma_gemm_f16(
    const h16* __restrict__ A,  int lda, ll sAo, ll sAi,
    const h16* __restrict__ Bw, int ldb, ll sBo, ll sBi,
    const float* __restrict__ bias,
    const float* __restrict__ Rres,
    void* __restrict__ Cp,      int ldc, ll sCo, ll sCi,
    int innerCount, int M, int N, int K, float alpha, int act, int outHalf)
{
    __shared__ __align__(16) h16 bT[2 * 64 * 32];   // 2 x 4 KB double buffer

    const int z     = blockIdx.z;
    const int outer = z / innerCount;
    const int inner = z % innerCount;

    const h16*  Ab = A  + outer * sAo + inner * sAi;
    const h16*  Bb = Bw + outer * sBo + inner * sBi;
    const ll    cOff = outer * sCo + inner * sCi;
    float*      Cf = (float*)Cp + cOff;
    h16*        Ch = (h16*)Cp + cOff;
    const float* Rb = Rres ? (Rres + cOff) : nullptr;

    const int tid    = threadIdx.x;
    const int lane   = tid & 31;
    const int wave   = tid >> 5;
    const int tile_m = blockIdx.x * 128 + wave * 32;
    const int tile_n = blockIdx.y * 64;
    const bool active = tile_m < M;                  // wave-uniform

    // B-tile fill mapping: thread t copies 16 halfs of column (t>>1).
    const int fcol = tid >> 1;                       // 0..63
    const int fseg = (tid & 1) << 4;                 // 0 or 16
    const int fcg  = tile_n + fcol;
    const bool fOK = fcg < N;
    const h16* fsrc = Bb + (ll)fcg * ldb + fseg;     // + k0 each step
    h16* fdst0 = &bT[fcol * 32 + fseg];
    h16* fdst1 = fdst0 + 2048;

    // Per-lane fragment mapping (contiguous k = 16*(lane/16) + s for both A and B).
    const int arow = lane & 15;
    const int aoff = (lane >> 4) << 4;               // 0 or 16 (A K-half)
    const int kh   = (lane >> 4) << 4;               // 0 or 16 (B K-half)
    const int cl   = lane & 15;                      // col within 16-tile

    const h16* Arow0 = Ab + (ll)(tile_m + arow) * lda + aoff;
    const h16* Arow1 = Arow0 + (ll)16 * lda;

    v8f acc0[4] = {{}, {}, {}, {}};
    v8f acc1[4] = {{}, {}, {}, {}};

    // Prologue: OOB columns stay zero forever; kick async fill of buffer 0.
    if (!fOK) {
        const uint4 zz = {0, 0, 0, 0};
        *(uint4*)fdst0 = zz;  *(uint4*)(fdst0 + 8) = zz;
        *(uint4*)fdst1 = zz;  *(uint4*)(fdst1 + 8) = zz;
    } else {
        copy16_g2l(fdst0, fsrc);
        copy16_g2l(fdst0 + 8, fsrc + 8);
    }

    const int steps = K >> 5;
    for (int i = 0; i < steps; ++i) {
        const int k0 = i << 5;
        wait_g2l();                   // own async fills of current buffer done
        __syncthreads();              // all fills done; all prior reads done

        // Kick fill of the other buffer for the next K-step (overlaps WMMAs).
        if (i + 1 < steps && fOK) {
            const h16* nsrc = fsrc + k0 + 32;
            h16* ndst = (i & 1) ? fdst0 : fdst1;
            copy16_g2l(ndst, nsrc);
            copy16_g2l(ndst + 8, nsrc + 8);
        }

        if (active) {
            const h16* bcur = bT + ((i & 1) ? 2048 : 0);

            Frag a0, a1;
            a0.q[0] = *(const uint4*)(Arow0 + k0);
            a0.q[1] = *(const uint4*)(Arow0 + k0 + 8);
            a1.q[0] = *(const uint4*)(Arow1 + k0);
            a1.q[1] = *(const uint4*)(Arow1 + k0 + 8);
            if (k0 + 32 < K)
                __builtin_prefetch((const void*)(Arow0 + k0 + 32), 0, 0);

#pragma unroll
            for (int s = 0; s < 4; ++s) {
                Frag bf;
                const h16* bp = bcur + (s * 16 + cl) * 32 + kh;
                bf.q[0] = *(const uint4*)bp;
                bf.q[1] = *(const uint4*)(bp + 8);
                acc0[s] = __builtin_amdgcn_wmma_f32_16x16x32_f16(
                    false, a0.v, false, bf.v, (short)0, acc0[s], false, false);
                acc1[s] = __builtin_amdgcn_wmma_f32_16x16x32_f16(
                    false, a1.v, false, bf.v, (short)0, acc1[s], false, false);
            }
        }
    }

    if (active) {
#pragma unroll
        for (int s = 0; s < 4; ++s) {
            const int col = tile_n + s * 16 + cl;
            if (col >= N) continue;
            const float bv = bias ? bias[col] : 0.0f;
#pragma unroll
            for (int r = 0; r < 2; ++r) {
                const int rbase = tile_m + r * 16 + ((lane >> 4) << 3);
#pragma unroll
                for (int j = 0; j < 8; ++j) {
                    const int row = rbase + j;          // C: VGPR j -> row j / j+8
                    const ll off = (ll)row * ldc + col;
                    float v = (r ? acc1[s][j] : acc0[s][j]) * alpha + bv;
                    if (act == 1) v = 0.5f * v * (1.0f + erff(v * 0.70710678118654752f));
                    if (Rb) v += Rb[off];
                    if (outHalf) Ch[off] = (h16)v;
                    else         Cf[off] = v;
                }
            }
        }
    }
}

// ---------------------------------------------------------------------------
// Elementwise fp32 -> f16 (weights, once per launch).
// ---------------------------------------------------------------------------
__global__ void cvt_f32_f16_kernel(const float* __restrict__ src,
                                   h16* __restrict__ dst, ll n)
{
    const ll i = (ll)blockIdx.x * blockDim.x + threadIdx.x;
    if (i < n) dst[i] = (h16)src[i];
}

// ---------------------------------------------------------------------------
// LayerNorm over rows of length Dn; fp32 in, f16 out (one block per row).
// ---------------------------------------------------------------------------
__launch_bounds__(256)
__global__ void layernorm_kernel(const float* __restrict__ X,
                                 const float* __restrict__ g,
                                 const float* __restrict__ b,
                                 h16* __restrict__ Y, int Dn)
{
    const ll row = blockIdx.x;
    const float* x = X + row * Dn;
    h16*         y = Y + row * Dn;
    const int tid = threadIdx.x, lane = tid & 31, wave = tid >> 5;

    float s = 0.0f, s2 = 0.0f;
    for (int i = tid; i < Dn; i += 256) { const float v = x[i]; s += v; s2 += v * v; }
#pragma unroll
    for (int o = 16; o > 0; o >>= 1) { s += __shfl_xor(s, o, 32); s2 += __shfl_xor(s2, o, 32); }

    __shared__ float rs[8], rs2[8];
    __shared__ float mu_s, r_s;
    if (lane == 0) { rs[wave] = s; rs2[wave] = s2; }
    __syncthreads();
    if (tid == 0) {
        float t = 0.0f, t2 = 0.0f;
        for (int i = 0; i < 8; ++i) { t += rs[i]; t2 += rs2[i]; }
        const float mu  = t / Dn;
        const float var = t2 / Dn - mu * mu;
        mu_s = mu;
        r_s  = rsqrtf(var + 1e-5f);
    }
    __syncthreads();
    const float mu = mu_s, r = r_s;
    for (int i = tid; i < Dn; i += 256)
        y[i] = (h16)((x[i] - mu) * r * g[i] + b[i]);
}

// ---------------------------------------------------------------------------
// Masked softmax over rows of 224 (valid keys < Tvalid).  Reads fp32 row and
// writes f16 probabilities IN PLACE into the same row region (first 448 bytes
// of the 896-byte fp32 row), i.e. probs = (h16*)scores with row stride 448.
// ---------------------------------------------------------------------------
__launch_bounds__(256)
__global__ void softmax_kernel(float* __restrict__ S, int Tvalid)
{
    const int Tp = 224;
    const ll row = blockIdx.x;
    float* x = S + row * Tp;
    h16*   y = (h16*)x;
    const int tid = threadIdx.x, lane = tid & 31, wave = tid >> 5;

    const float v = (tid < Tvalid) ? x[tid] : -3.0e38f;
    float m = v;
#pragma unroll
    for (int o = 16; o > 0; o >>= 1) m = fmaxf(m, __shfl_xor(m, o, 32));

    __shared__ float red[8];
    __shared__ float bm, bs;
    if (lane == 0) red[wave] = m;
    __syncthreads();
    if (tid == 0) {
        float t = red[0];
        for (int i = 1; i < 8; ++i) t = fmaxf(t, red[i]);
        bm = t;
    }
    __syncthreads();

    const float e = (tid < Tvalid) ? __expf(v - bm) : 0.0f;
    float s = e;
#pragma unroll
    for (int o = 16; o > 0; o >>= 1) s += __shfl_xor(s, o, 32);
    if (lane == 0) red[wave] = s;   // safe: barrier above already passed
    __syncthreads();
    if (tid == 0) {
        float t = 0.0f;
        for (int i = 0; i < 8; ++i) t += red[i];
        bs = 1.0f / t;
    }
    __syncthreads();                 // all fp32 reads done before f16 writes
    if (tid < Tp) y[tid] = (h16)(e * bs);
}

// ---------------------------------------------------------------------------
// im2col: x(B,3,224,224) fp32 -> P(B*196, 768) f16, k = py*48 + px*3 + c.
// ---------------------------------------------------------------------------
__global__ void im2col_kernel(const float* __restrict__ x, h16* __restrict__ P)
{
    const ll idx = (ll)blockIdx.x * blockDim.x + threadIdx.x;
    if (idx >= 6272LL * 768) return;
    const int k = (int)(idx % 768);
    const ll  m = idx / 768;
    const int b = (int)(m / 196), n = (int)(m % 196);
    const int hp = n / 14, wp = n % 14;
    const int py = k / 48, rem = k % 48, px = rem / 3, c = rem % 3;
    P[idx] = (h16)x[(((ll)b * 3 + c) * 224 + (hp * 16 + py)) * 224 + (wp * 16 + px)];
}

// ---------------------------------------------------------------------------
// Assemble h(B,224,768) fp32: t=0 cls+pos, 1..196 emb+pos, >=197 zero pad.
// ---------------------------------------------------------------------------
__global__ void assemble_kernel(const float* __restrict__ emb,
                                const float* __restrict__ cls,
                                const float* __restrict__ pos,
                                float* __restrict__ h)
{
    const ll idx = (ll)blockIdx.x * blockDim.x + threadIdx.x;
    if (idx >= 32LL * 224 * 768) return;
    const int d = (int)(idx % 768);
    const ll  r = idx / 768;
    const int t = (int)(r % 224);
    const int b = (int)(r / 224);
    float v;
    if (t == 0)        v = cls[d] + pos[d];
    else if (t <= 196) v = emb[((ll)b * 196 + (t - 1)) * 768 + d] + pos[(ll)t * 768 + d];
    else               v = 0.0f;
    h[idx] = v;
}

// ---------------------------------------------------------------------------
// V transpose: qkvh(z,224,192)[...,128:192] f16 -> vT(z,64,224) f16.
// ---------------------------------------------------------------------------
__global__ void vtrans_kernel(const h16* __restrict__ qkvh, h16* __restrict__ vT)
{
    const ll idx = (ll)blockIdx.x * blockDim.x + threadIdx.x;
    if (idx >= 384LL * 64 * 224) return;
    const int t = (int)(idx % 224);
    const ll  r = idx / 224;
    const int d = (int)(r % 64);
    const int z = (int)(r / 64);
    vT[idx] = qkvh[(ll)z * 43008 + (ll)t * 192 + 128 + d];
}

// ---------------------------------------------------------------------------
// Gather CLS rows of final h into f16 (32 x 768).
// ---------------------------------------------------------------------------
__global__ void cls_gather_kernel(const float* __restrict__ h, h16* __restrict__ clsh)
{
    const int idx = blockIdx.x * blockDim.x + threadIdx.x;
    if (idx >= 32 * 768) return;
    const int d = idx % 768, b = idx / 768;
    clsh[idx] = (h16)h[(ll)b * 172032 + d];
}

// ---------------------------------------------------------------------------
// Host-side orchestration.
// ---------------------------------------------------------------------------
extern "C" void kernel_launch(void* const* d_in, const int* in_sizes, int n_in,
                              void* d_out, int out_size, void* d_ws, size_t ws_size,
                              hipStream_t stream)
{
    const float* x       = (const float*)d_in[0];
    const float* patch_w = (const float*)d_in[1];
    const float* patch_b = (const float*)d_in[2];
    const float* pos_enc = (const float*)d_in[3];
    const float* cls_tok = (const float*)d_in[4];
    const float* qkv_w   = (const float*)d_in[5];
    const float* qkv_b   = (const float*)d_in[6];
    const float* ln_g    = (const float*)d_in[7];
    const float* ln_b    = (const float*)d_in[8];
    const float* w1      = (const float*)d_in[9];
    const float* b1      = (const float*)d_in[10];
    const float* w2      = (const float*)d_in[11];
    const float* b2      = (const float*)d_in[12];
    const float* head_w  = (const float*)d_in[13];
    const float* head_b  = (const float*)d_in[14];
    float* out = (float*)d_out;

    const int D = 768, Hh = 12, Lc = 12, F = 3072, Tp = 224;
    const ll rowsBT = 32LL * Tp;             // 7168
    const ll sBatch = (ll)Tp * D;            // 172032

    // ---- workspace carve (all offsets 256B aligned) ----
    char* wsb = (char*)d_ws;
    float* h      = (float*)wsb;                       wsb += 22020096;   // 32*224*768 f32
    float* scores = (float*)wsb;                       wsb += 77070336;   // 384*224*224 f32 (probs alias, f16 stride 448)
    h16* hn       = (h16*)wsb;                         wsb += 11010048;   // 32*224*768 f16
    h16* qkvh     = (h16*)wsb;                         wsb += 33030144;   // 384*224*192 f16
    h16* vT       = (h16*)wsb;                         wsb += 11010048;   // 384*64*224 f16
    h16* u16      = (h16*)wsb;                         wsb += 44040192;   // 7168*3072 f16
    h16* clsh     = (h16*)wsb;                         wsb += 49152;      // 32*768 f16
    h16* pw16     = (h16*)wsb;                         wsb += 1179648;    // 768*768
    h16* qw16     = (h16*)wsb;                         wsb += 294912;     // 12*192*64
    h16* w1_16    = (h16*)wsb;                         wsb += 56623104;   // 12*3072*768
    h16* w2_16    = (h16*)wsb;                         wsb += 56623104;   // 12*768*3072
    h16* hw16     = (h16*)wsb;                         wsb += 1536000;    // 1000*768
    h16* Ph       = u16;                               // alias: patch im2col (pre-layers)
    float* emb    = scores;                            // alias: patch GEMM out (pre-layers)
    h16* probs    = (h16*)scores;                      // f16 probs, row stride 448 halfs

    const dim3 blk(128);
    auto cdiv = [](ll a, ll b) { return (unsigned)((a + b - 1) / b); };

    // ---- one-time fp32 -> f16 weight conversion ----
    {
        struct { const float* s; h16* d; ll n; } cv[5] = {
            {patch_w, pw16,  589824}, {qkv_w, qw16, 147456},
            {w1, w1_16, 28311552},    {w2, w2_16, 28311552},
            {head_w, hw16, 768000}};
        for (int i = 0; i < 5; ++i)
            cvt_f32_f16_kernel<<<cdiv(cv[i].n, 256), 256, 0, stream>>>(cv[i].s, cv[i].d, cv[i].n);
    }

    // ---- patch embedding ----
    im2col_kernel<<<cdiv(6272LL * 768, 256), 256, 0, stream>>>(x, Ph);
    wmma_gemm_f16<<<dim3(6272 / 128, 768 / 64, 1), blk, 0, stream>>>(
        Ph, 768, 0, 0,  pw16, 768, 0, 0,
        patch_b, nullptr, emb, 768, 0, 0,
        1, 6272, 768, 768, 1.0f, 0, 0);
    assemble_kernel<<<cdiv(32LL * 224 * 768, 256), 256, 0, stream>>>(emb, cls_tok, pos_enc, h);

    // ---- transformer layers ----
    for (int l = 0; l < Lc; ++l) {
        const float* lg  = ln_g + (ll)l * D;
        const float* lb  = ln_b + (ll)l * D;
        const float* qb  = qkv_b + (ll)l * 192;
        const float* b1l = b1 + (ll)l * F;
        const float* b2l = b2 + (ll)l * D;
        const h16* qwl = qw16 + (ll)l * 192 * 64;
        const h16* w1l = w1_16 + (ll)l * F * D;
        const h16* w2l = w2_16 + (ll)l * D * F;

        // LN1 (fp32 -> f16)
        layernorm_kernel<<<dim3((unsigned)rowsBT), 256, 0, stream>>>(h, lg, lb, hn, D);

        // QKV (f16 out): per (b,head) (224x64)@(192x64)^T -> (224x192)
        wmma_gemm_f16<<<dim3(cdiv(Tp, 128), 192 / 64, 384), blk, 0, stream>>>(
            hn, D, sBatch, 64,
            qwl, 64, 0, 0,
            qb, nullptr,
            qkvh, 192, (ll)Hh * Tp * 192, (ll)Tp * 192,
            Hh, Tp, 192, 64, 1.0f, 0, 1);

        // V transpose (f16): (224x64) -> (64x224) per z
        vtrans_kernel<<<cdiv(384LL * 64 * 224, 256), 256, 0, stream>>>(qkvh, vT);

        // scores (fp32) = (Q @ K^T)/8 : per z (224x64)@(224x64)^T
        wmma_gemm_f16<<<dim3(cdiv(Tp, 128), cdiv(Tp, 64), 384), blk, 0, stream>>>(
            qkvh,      192, (ll)Tp * 192, 0,
            qkvh + 64, 192, (ll)Tp * 192, 0,
            nullptr, nullptr,
            scores, Tp, (ll)Tp * Tp, 0,
            1, Tp, Tp, 64, 0.125f, 0, 0);

        // masked softmax -> f16 probs in place (row stride 448 halfs)
        softmax_kernel<<<dim3(384 * Tp), 256, 0, stream>>>(scores, 197);

        // h += P @ V : per (b,head) (224x224)@(64x224)^T -> h[:, head*64:+64]
        wmma_gemm_f16<<<dim3(cdiv(Tp, 128), 1, 384), blk, 0, stream>>>(
            probs, 448, (ll)Hh * Tp * 448, (ll)Tp * 448,
            vT,    224, (ll)Hh * 64 * 224, (ll)64 * 224,
            nullptr, h,
            h, D, sBatch, 64,
            Hh, Tp, 64, Tp, 1.0f, 0, 0);

        // LN2 (same gamma/beta per reference)
        layernorm_kernel<<<dim3((unsigned)rowsBT), 256, 0, stream>>>(h, lg, lb, hn, D);

        // MLP1 + exact GELU (f16 out): (7168x768)@(3072x768)^T -> u16
        wmma_gemm_f16<<<dim3((unsigned)(rowsBT / 128), F / 64, 1), blk, 0, stream>>>(
            hn, D, 0, 0,
            w1l, D, 0, 0,
            b1l, nullptr,
            u16, F, 0, 0,
            1, (int)rowsBT, F, D, 1.0f, 1, 1);

        // MLP2 + residual (fp32): h = u @ (768x3072)^T + b2 + h
        wmma_gemm_f16<<<dim3((unsigned)(rowsBT / 128), D / 64, 1), blk, 0, stream>>>(
            u16, F, 0, 0,
            w2l, F, 0, 0,
            b2l, h,
            h, D, 0, 0,
            1, (int)rowsBT, D, F, 1.0f, 0, 0);
    }

    // ---- head: out(32,1000) = h[:,0,:] @ head_w^T + head_b ----
    cls_gather_kernel<<<cdiv(32 * 768, 256), 256, 0, stream>>>(h, clsh);
    wmma_gemm_f16<<<dim3(1, cdiv(1000, 64), 1), blk, 0, stream>>>(
        clsh, 768, 0, 0,
        hw16, 768, 0, 0,
        head_b, nullptr,
        out, 1000, 0, 0,
        1, 32, 1000, 768, 1.0f, 0, 0);
}
